// EEGfMRIBridgeFusionNet_80977313399249
// MI455X (gfx1250) — matrix operand order repaired and checked by reading, not compiled, not run on hardware
//
#include <hip/hip_runtime.h>
#include <hip/hip_bf16.h>

typedef _Float16 half_t;
typedef __attribute__((ext_vector_type(16))) _Float16 v16h;
typedef __attribute__((ext_vector_type(8)))  float    v8f;

#define BD     128
#define PITCH  128
#define LN_EPS 1e-5f

// ---------------- fragment helpers (gfx1250 wave32 WMMA layouts) ----------------

union AFrag { v16h v; half_t h[16]; unsigned int u[8]; };
union BFrag { v16h v; uint4 q[2]; };

// A 16x32 f16: lane L -> row m = L%16 ; VGPR p holds K pair KB[p]+16*(L/16), +1
__device__ inline void load_a_lds(AFrag& a, const half_t* buf, int m_a, int g, int k0) {
    constexpr int KB[8] = {0, 2, 4, 6, 8, 10, 12, 14};
#pragma unroll
    for (int p = 0; p < 8; ++p) {
        int k = k0 + KB[p] + 16 * g;
        a.u[p] = *reinterpret_cast<const unsigned int*>(buf + m_a * PITCH + k);
    }
}

__device__ inline void load_a_glob(AFrag& a, const float* src, int width, int row, int g, int k0) {
    constexpr int KB[8] = {0, 2, 4, 6, 8, 10, 12, 14};
#pragma unroll
    for (int p = 0; p < 8; ++p) {
        int k = k0 + KB[p] + 16 * g;
        float2 f = *reinterpret_cast<const float2*>(src + (size_t)row * width + k);
        a.h[2 * p]     = (half_t)f.x;
        a.h[2 * p + 1] = (half_t)f.y;
    }
}

// B 32x16 f16 from W^T stored [N][K] row-major: lane L -> col n = 16*jn + L%16,
// 16 contiguous halves K = k0 + 16*(L/16) .. +15  => two b128 loads.
__device__ inline void load_b(BFrag& b, const half_t* wt, int Kw, int nl, int g, int jn, int k0) {
    const uint4* p = reinterpret_cast<const uint4*>(wt + (size_t)(16 * jn + nl) * Kw + k0 + 16 * g);
    b.q[0] = p[0];
    b.q[1] = p[1];
}

__device__ inline v8f wmma_step(const AFrag& a, const BFrag& b, v8f c) {
    return __builtin_amdgcn_wmma_f32_16x16x32_f16(false, a.v, false, b.v, (short)0, c, false, false);
}

// C layout: lane L holds (m = r + 8*(L/16), n = 16*j + L%16), r = 0..7
template <int NT>
__device__ inline void store_c(const v8f acc[NT], half_t* buf, int nl, int g) {
#pragma unroll
    for (int j = 0; j < NT; ++j)
#pragma unroll
        for (int r = 0; r < 8; ++r)
            buf[(r + 8 * g) * PITCH + 16 * j + nl] = (half_t)acc[j][r];
}

template <int NT, int KS>
__device__ inline void gemm_lds(v8f acc[NT], const half_t* abuf, const half_t* wt, int Kw,
                                const float* bias, int nl, int g) {
    AFrag af[KS];
#pragma unroll
    for (int kk = 0; kk < KS; ++kk) load_a_lds(af[kk], abuf, nl, g, 32 * kk);
#pragma unroll
    for (int j = 0; j < NT; ++j) {
        float bv = bias[16 * j + nl];
        v8f c;
#pragma unroll
        for (int r = 0; r < 8; ++r) c[r] = bv;
#pragma unroll
        for (int kk = 0; kk < KS; ++kk) {
            BFrag bf;
            load_b(bf, wt, Kw, nl, g, j, 32 * kk);
            c = wmma_step(af[kk], bf, c);
        }
        acc[j] = c;
    }
}

template <int NT, int KS>
__device__ inline void gemm_glob(v8f acc[NT], const float* x, int width, int row,
                                 const half_t* wt, int Kw, const float* bias, int nl, int g) {
    AFrag af[KS];
#pragma unroll
    for (int kk = 0; kk < KS; ++kk) load_a_glob(af[kk], x, width, row, g, 32 * kk);
#pragma unroll
    for (int j = 0; j < NT; ++j) {
        float bv = bias[16 * j + nl];
        v8f c;
#pragma unroll
        for (int r = 0; r < 8; ++r) c[r] = bv;
#pragma unroll
        for (int kk = 0; kk < KS; ++kk) {
            BFrag bf;
            load_b(bf, wt, Kw, nl, g, j, 32 * kk);
            c = wmma_step(af[kk], bf, c);
        }
        acc[j] = c;
    }
}

// LayerNorm + activation across feature dim = 16*NT, rows spread over 16-lane half-wave
template <int NT, bool GELU>
__device__ inline void ln_act(v8f acc[NT], const float* gam, const float* bet, int nl) {
    const float invD = 1.0f / (16.0f * NT);
#pragma unroll
    for (int r = 0; r < 8; ++r) {
        float s = 0.f, s2 = 0.f;
#pragma unroll
        for (int j = 0; j < NT; ++j) { float x = acc[j][r]; s += x; s2 += x * x; }
#pragma unroll
        for (int off = 1; off < 16; off <<= 1) {
            s  += __shfl_xor(s,  off, 32);
            s2 += __shfl_xor(s2, off, 32);
        }
        float mean = s * invD;
        float var  = s2 * invD - mean * mean;
        float rstd = rsqrtf(var + LN_EPS);
#pragma unroll
        for (int j = 0; j < NT; ++j) {
            int n   = 16 * j + nl;
            float x = (acc[j][r] - mean) * rstd * gam[n] + bet[n];
            acc[j][r] = GELU ? 0.5f * x * (1.0f + erff(x * 0.70710678118f)) : fmaxf(x, 0.f);
        }
    }
}

// ---------------- phase 0: weight transpose+convert f32 [K][N] -> f16 [N][K] ----------------

__global__ void transpose_to_f16(const float* __restrict__ src, half_t* __restrict__ dst,
                                 int K, int N) {
    int i = blockIdx.x * blockDim.x + threadIdx.x;
    if (i >= K * N) return;
    int k = i / N, n = i % N;
    dst[(size_t)n * K + k] = (half_t)src[i];
}

__global__ void pack_wc2(const float* __restrict__ src, half_t* __restrict__ dst) {
    // Wc2 [64][2] f32 -> padded f16 [16][64] (rows >= 2 zero)
    int i = blockIdx.x * blockDim.x + threadIdx.x;
    if (i >= 16 * 64) return;
    int n = i / 64, k = i % 64;
    dst[i] = (n < 2) ? (half_t)src[k * 2 + n] : (half_t)0.f;
}

// ---------------- main fused kernel: 2 waves/block, 16 batch rows per wave ----------------

struct WaveBufs {
    half_t E[16 * PITCH];   // eeg_p -> ctx -> h
    half_t F[16 * PITCH];   // fmri_p (kept until fusion)
    half_t Q[16 * PITCH];   // q -> fused
    half_t K0[16 * PITCH];
    half_t K1[16 * PITCH];
    half_t V0[16 * PITCH];
    half_t V1[16 * PITCH];
};

struct Smem {
    WaveBufs wb[2];
    float c_be[128], c_ge[128], c_bge[128];
    float c_bf[128], c_gf[128], c_bgf[128];
    float c_bq[128], c_bk[128], c_bv[128], c_bo[128];
    float c_bc1[64], c_gc[64], c_bgc[64], c_bc2[16];
};

__global__ void __launch_bounds__(64) fusion_fwd(
    const float* __restrict__ eeg, const float* __restrict__ fmri,
    const float* __restrict__ be_, const float* __restrict__ ge_, const float* __restrict__ bge_,
    const float* __restrict__ bf_, const float* __restrict__ gf_, const float* __restrict__ bgf_,
    const float* __restrict__ bq_, const float* __restrict__ bk_,
    const float* __restrict__ bv_, const float* __restrict__ bo_,
    const float* __restrict__ flog, const float* __restrict__ temp,
    const float* __restrict__ bc1_, const float* __restrict__ gc_,
    const float* __restrict__ bgc_, const float* __restrict__ bc2_,
    const half_t* __restrict__ WeT, const half_t* __restrict__ WfT,
    const half_t* __restrict__ WqT, const half_t* __restrict__ WkT,
    const half_t* __restrict__ WvT, const half_t* __restrict__ WoT,
    const half_t* __restrict__ Wc1T, const half_t* __restrict__ Wc2T,
    float* __restrict__ out, int nTiles)
{
    __shared__ Smem sm;
    const int tid = threadIdx.x;

    // stage small constants into LDS once per block
    for (int i = tid; i < 128; i += 64) {
        sm.c_be[i] = be_[i]; sm.c_ge[i] = ge_[i]; sm.c_bge[i] = bge_[i];
        sm.c_bf[i] = bf_[i]; sm.c_gf[i] = gf_[i]; sm.c_bgf[i] = bgf_[i];
        sm.c_bq[i] = bq_[i]; sm.c_bk[i] = bk_[i];
        sm.c_bv[i] = bv_[i]; sm.c_bo[i] = bo_[i];
    }
    if (tid < 64) { sm.c_bc1[tid] = bc1_[tid]; sm.c_gc[tid] = gc_[tid]; sm.c_bgc[tid] = bgc_[tid]; }
    if (tid < 16) sm.c_bc2[tid] = (tid < 2) ? bc2_[tid] : 0.f;
    __syncthreads();

    const int wave = tid >> 5;
    const int lane = tid & 31;
    const int g    = lane >> 4;    // half-wave group
    const int nl   = lane & 15;    // lane-within-group (== A-row, == C/B column-in-tile)
    WaveBufs& wb = sm.wb[wave];

    const int tile   = blockIdx.x * 2 + wave;
    const bool active = tile < nTiles;
    const int row0   = (active ? tile : nTiles - 1) * 16;

    v8f acc[8];

    // ---- modality projections: Linear + LN + GELU ----
    gemm_glob<8, 4>(acc, eeg, 128, row0 + nl, WeT, 128, sm.c_be, nl, g);
    ln_act<8, true>(acc, sm.c_ge, sm.c_bge, nl);
    store_c<8>(acc, wb.E, nl, g);

    gemm_glob<8, 2>(acc, fmri, 64, row0 + nl, WfT, 64, sm.c_bf, nl, g);
    ln_act<8, true>(acc, sm.c_gf, sm.c_bgf, nl);
    store_c<8>(acc, wb.F, nl, g);
    __syncthreads();

    // ---- Q / K / V projections ----
    gemm_lds<8, 4>(acc, wb.E, WqT, 128, sm.c_bq, nl, g); store_c<8>(acc, wb.Q,  nl, g);
    gemm_lds<8, 4>(acc, wb.E, WkT, 128, sm.c_bk, nl, g); store_c<8>(acc, wb.K0, nl, g);
    gemm_lds<8, 4>(acc, wb.F, WkT, 128, sm.c_bk, nl, g); store_c<8>(acc, wb.K1, nl, g);
    gemm_lds<8, 4>(acc, wb.E, WvT, 128, sm.c_bv, nl, g); store_c<8>(acc, wb.V0, nl, g);
    gemm_lds<8, 4>(acc, wb.F, WvT, 128, sm.c_bv, nl, g); store_c<8>(acc, wb.V1, nl, g);
    __syncthreads();

    // ---- 2-token, 4-head attention: 2 (row,head) pairs per lane; ctx -> wb.E ----
#pragma unroll
    for (int p = 0; p < 2; ++p) {
        int idx = lane * 2 + p;
        int m = idx >> 2, h = idx & 3;
        const half_t* qr  = wb.Q  + m * PITCH + h * 32;
        const half_t* k0r = wb.K0 + m * PITCH + h * 32;
        const half_t* k1r = wb.K1 + m * PITCH + h * 32;
        const half_t* v0r = wb.V0 + m * PITCH + h * 32;
        const half_t* v1r = wb.V1 + m * PITCH + h * 32;
        float s0 = 0.f, s1 = 0.f;
#pragma unroll
        for (int d = 0; d < 32; ++d) {
            float qv = (float)qr[d];
            s0 += qv * (float)k0r[d];
            s1 += qv * (float)k1r[d];
        }
        s0 *= 0.17677669529f;   // 1/sqrt(32)
        s1 *= 0.17677669529f;
        float mx = fmaxf(s0, s1);
        float e0 = __expf(s0 - mx), e1 = __expf(s1 - mx);
        float inv = 1.f / (e0 + e1);
        float a0 = e0 * inv, a1 = e1 * inv;
        half_t* cx = wb.E + m * PITCH + h * 32;
#pragma unroll
        for (int d = 0; d < 32; ++d)
            cx[d] = (half_t)(a0 * (float)v0r[d] + a1 * (float)v1r[d]);
    }
    __syncthreads();

    // ---- output projection + learned fusion; fused -> wb.Q ----
    gemm_lds<8, 4>(acc, wb.E, WoT, 128, sm.c_bo, nl, g);
    {
        float T  = temp[0];
        float l0 = flog[0] / T, l1 = flog[1] / T;
        float mx = fmaxf(l0, l1);
        float e0 = __expf(l0 - mx), e1 = __expf(l1 - mx);
        float inv = 1.f / (e0 + e1);
        float w0 = e0 * inv, w1 = e1 * inv;
#pragma unroll
        for (int j = 0; j < 8; ++j)
#pragma unroll
            for (int r = 0; r < 8; ++r) {
                float fp = (float)wb.F[(r + 8 * g) * PITCH + 16 * j + nl];
                acc[j][r] = w0 * acc[j][r] + w1 * fp;
            }
    }
    store_c<8>(acc, wb.Q, nl, g);
    __syncthreads();

    // ---- classifier: Linear(128->64) + LN + ReLU ----
    v8f acc4[4];
    gemm_lds<4, 4>(acc4, wb.Q, Wc1T, 128, sm.c_bc1, nl, g);
    ln_act<4, false>(acc4, sm.c_gc, sm.c_bgc, nl);
    store_c<4>(acc4, wb.E, nl, g);
    __syncthreads();

    // ---- final Linear(64->2) via zero-padded WMMA tile ----
    v8f accO[1];
    gemm_lds<1, 2>(accO, wb.E, Wc2T, 64, sm.c_bc2, nl, g);
    if (active && nl < 2) {
#pragma unroll
        for (int r = 0; r < 8; ++r)
            out[(size_t)(row0 + r + 8 * g) * 2 + nl] = accO[0][r];
    }
}

// ---------------- host launch ----------------

extern "C" void kernel_launch(void* const* d_in, const int* in_sizes, int n_in,
                              void* d_out, int out_size, void* d_ws, size_t ws_size,
                              hipStream_t stream) {
    const float* eeg  = (const float*)d_in[0];
    const float* fmri = (const float*)d_in[1];
    const float* We   = (const float*)d_in[2];
    const float* be   = (const float*)d_in[3];
    const float* ge   = (const float*)d_in[4];
    const float* bge  = (const float*)d_in[5];
    const float* Wf   = (const float*)d_in[6];
    const float* bf   = (const float*)d_in[7];
    const float* gf   = (const float*)d_in[8];
    const float* bgf  = (const float*)d_in[9];
    const float* Wq   = (const float*)d_in[10];
    const float* bq   = (const float*)d_in[11];
    const float* Wk   = (const float*)d_in[12];
    const float* bk   = (const float*)d_in[13];
    const float* Wv   = (const float*)d_in[14];
    const float* bv   = (const float*)d_in[15];
    const float* Wo   = (const float*)d_in[16];
    const float* bo   = (const float*)d_in[17];
    const float* flog = (const float*)d_in[18];
    const float* temp = (const float*)d_in[19];
    const float* Wc1  = (const float*)d_in[20];
    const float* bc1  = (const float*)d_in[21];
    const float* gc   = (const float*)d_in[22];
    const float* bgc  = (const float*)d_in[23];
    const float* Wc2  = (const float*)d_in[24];
    const float* bc2  = (const float*)d_in[25];

    const int Btot   = in_sizes[0] / 128;   // eeg_feats is [B,128]
    const int nTiles = Btot / 16;

    // ws layout (f16 elements): transposed weights
    half_t* ws   = (half_t*)d_ws;
    half_t* WeT  = ws;                 // [128][128]
    half_t* WfT  = WeT  + 128 * 128;   // [128][64]
    half_t* WqT  = WfT  + 128 * 64;    // [128][128]
    half_t* WkT  = WqT  + 128 * 128;
    half_t* WvT  = WkT  + 128 * 128;
    half_t* WoT  = WvT  + 128 * 128;
    half_t* Wc1T = WoT  + 128 * 128;   // [64][128]
    half_t* Wc2T = Wc1T + 64 * 128;    // [16][64] padded

    transpose_to_f16<<<(128 * 128 + 255) / 256, 256, 0, stream>>>(We,  WeT,  128, 128);
    transpose_to_f16<<<(64  * 128 + 255) / 256, 256, 0, stream>>>(Wf,  WfT,  64,  128);
    transpose_to_f16<<<(128 * 128 + 255) / 256, 256, 0, stream>>>(Wq,  WqT,  128, 128);
    transpose_to_f16<<<(128 * 128 + 255) / 256, 256, 0, stream>>>(Wk,  WkT,  128, 128);
    transpose_to_f16<<<(128 * 128 + 255) / 256, 256, 0, stream>>>(Wv,  WvT,  128, 128);
    transpose_to_f16<<<(128 * 128 + 255) / 256, 256, 0, stream>>>(Wo,  WoT,  128, 128);
    transpose_to_f16<<<(128 * 64  + 255) / 256, 256, 0, stream>>>(Wc1, Wc1T, 128, 64);
    pack_wc2<<<(16 * 64 + 255) / 256, 256, 0, stream>>>(Wc2, Wc2T);

    const int blocks = (nTiles + 1) / 2;   // 2 waves (tiles) per 64-thread block
    fusion_fwd<<<blocks, 64, 0, stream>>>(
        eeg, fmri, be, ge, bge, bf, gf, bgf, bq, bk, bv, bo, flog, temp,
        bc1, gc, bgc, bc2,
        WeT, WfT, WqT, WkT, WvT, WoT, Wc1T, Wc2T,
        (float*)d_out, nTiles);
}
